// GAT_17360257810577
// MI455X (gfx1250) — matrix operand order repaired and checked by reading, not compiled, not run on hardware
//
#include <hip/hip_runtime.h>
#include <hip/hip_bf16.h>
#include <math.h>

typedef __attribute__((ext_vector_type(16))) _Float16 v16h;
typedef __attribute__((ext_vector_type(8)))  _Float16 v8h;
typedef __attribute__((ext_vector_type(4)))  _Float16 half4;
typedef __attribute__((ext_vector_type(8)))  float    v8f;

#define F_IN   512
#define HEADS1 8
#define D1     8
#define F_HID  64   // HEADS1*D1
#define C_OUT  40
#define NEG_SLOPE 0.2f

// ---------------------------------------------------------------- utilities
__global__ void k_fill(float* __restrict__ p, float v, long long n) {
    long long i = (long long)blockIdx.x * blockDim.x + threadIdx.x;
    if (i < n) p[i] = v;
}

// Pack W1 [512x64] f32 into WMMA B-fragment order, f16:
// W1p[((kt*64 + col)*32) + j] = W1[(kt*32 + ((j&7)|((j>>4)<<3)|((j&8)<<1)))*64 + col],  kt=0..15, col=0..63, j=0..31
// A lane needing (kt, col, half) reads 16 contiguous halves at offset hi*16.
__global__ void k_pack_w1(const float* __restrict__ W1, _Float16* __restrict__ W1p, int n) {
    int i = blockIdx.x * blockDim.x + threadIdx.x;       // over 16*64*32 = 32768
    if (i >= n) return;
    int j   = i & 31;
    int col = (i >> 5) & 63;
    int kt  = i >> 11;
    W1p[i] = (_Float16)W1[(size_t)(kt * 32 + ((j & 7) | ((j >> 4) << 3) | ((j & 8) << 1))) * F_HID + col];
}

__device__ __forceinline__ void edge_sd(const int* __restrict__ eidx, int E, int ei,
                                        int& s, int& d) {
    if (ei < E) { s = eidx[ei]; d = eidx[E + ei]; }
    else        { s = ei - E; d = ei - E; }              // self loop
}

__device__ __forceinline__ float lrelu(float v) { return v > 0.0f ? v : NEG_SLOPE * v; }

// ---------------------------------------------------------------- GEMM1: h1 = x @ W1 (WMMA f16->f32)
// block = 128 threads (4 waves). Each block: 32 rows x 64 cols of h1.
// Wave w owns column tile [w*16, w*16+16) and two row tiles (M=0..15, 16..31),
// sharing one B fragment per k-step between the two WMMAs.
__global__ __launch_bounds__(128) void k_gemm1_wmma(const float* __restrict__ x,
                                                    const _Float16* __restrict__ W1p,
                                                    float* __restrict__ h1, int N) {
    __shared__ _Float16 As[32][520];                     // 32 rows x 512 K, pitch 520 (bank-spread)
    const int rowBase = blockIdx.x * 32;
    const int tid = threadIdx.x;

    // Stage x tile -> LDS as f16. 32*512 f32 = 4096 float4; 128 threads x 32 iters, coalesced.
    for (int it = 0; it < 32; ++it) {
        int v  = it * 128 + tid;
        int r  = v >> 7;                                 // float4-index / 128 per row
        int c4 = v & 127;
        int row = rowBase + r; if (row >= N) row = N - 1;
        const float4 f = *(const float4*)(x + (size_t)row * F_IN + c4 * 4);
        half4 h; h.x = (_Float16)f.x; h.y = (_Float16)f.y;
                 h.z = (_Float16)f.z; h.w = (_Float16)f.w;
        *(half4*)&As[r][c4 * 4] = h;
    }
    __syncthreads();

    const int wave = tid >> 5;
    const int lane = tid & 31;
    const int mrow = lane & 15;
    const bool hi  = lane >= 16;
    const int col  = wave * 16 + mrow;                   // output column

    v8f acc0 = {}, acc1 = {};
    #pragma unroll 4
    for (int kt = 0; kt < 16; ++kt) {
        const int k0 = kt * 32;
        // B fragment: one contiguous 32B load from the packed, L2-resident buffer.
        v16h b = *(const v16h*)&W1p[((size_t)kt * 64 + col) * 32 + (hi ? 16 : 0)];

        // A fragments (16-bit A 16x32 layout):
        // lanes 0-15: elems 0-7 <-> K+0..7,  8-15 <-> K+16..23
        // lanes16-31: elems 0-7 <-> K+8..15, 8-15 <-> K+24..31
        v8h a0l = *(const v8h*)&As[mrow][k0 + (hi ? 8 : 0)];
        v8h a0h = *(const v8h*)&As[mrow][k0 + (hi ? 24 : 16)];
        v16h a0 = __builtin_shufflevector(a0l, a0h,
                  0,1,2,3,4,5,6,7, 8,9,10,11,12,13,14,15);
        acc0 = __builtin_amdgcn_wmma_f32_16x16x32_f16(false, a0, false, b,
                                                      (short)0, acc0, false, false);

        v8h a1l = *(const v8h*)&As[16 + mrow][k0 + (hi ? 8 : 0)];
        v8h a1h = *(const v8h*)&As[16 + mrow][k0 + (hi ? 24 : 16)];
        v16h a1 = __builtin_shufflevector(a1l, a1h,
                  0,1,2,3,4,5,6,7, 8,9,10,11,12,13,14,15);
        acc1 = __builtin_amdgcn_wmma_f32_16x16x32_f16(false, a1, false, b,
                                                      (short)0, acc1, false, false);
    }

    // C/D layout: VGPR i -> M = i (lanes 0-15) / i+8 (lanes 16-31), N = lane&15.
    if (rowBase + 32 <= N) {                             // block-uniform fast path: no branches
        #pragma unroll
        for (int i = 0; i < 8; ++i) {
            int r0 = rowBase + i + (hi ? 8 : 0);
            h1[(size_t)r0 * F_HID + col]        = acc0[i];
            h1[(size_t)(r0 + 16) * F_HID + col] = acc1[i];
        }
    } else {
        for (int i = 0; i < 8; ++i) {
            int r0 = rowBase + i + (hi ? 8 : 0);
            if (r0 < N)      h1[(size_t)r0 * F_HID + col]        = acc0[i];
            if (r0 + 16 < N) h1[(size_t)(r0 + 16) * F_HID + col] = acc1[i];
        }
    }
}

// ---------------------------------------------------------------- layer-1 attention scalars
__global__ void k_alpha1(const float* __restrict__ h1,
                         const float* __restrict__ att_s, const float* __restrict__ att_d,
                         float* __restrict__ as1, float* __restrict__ ad1, int N) {
    int i = blockIdx.x * blockDim.x + threadIdx.x;       // over N*HEADS1
    if (i >= N * HEADS1) return;
    int n = i >> 3, h = i & 7;
    const float* row = h1 + (size_t)n * F_HID + h * D1;
    float s = 0.f, d = 0.f;
    #pragma unroll
    for (int j = 0; j < D1; ++j) {
        s += row[j] * att_s[h * D1 + j];
        d += row[j] * att_d[h * D1 + j];
    }
    as1[i] = s; ad1[i] = d;
}

__global__ void k_edge_max1(const int* __restrict__ eidx, int E, int ET, int N,
                            const float* __restrict__ as1, const float* __restrict__ ad1,
                            float* __restrict__ mx1) {
    long long t = (long long)blockIdx.x * blockDim.x + threadIdx.x;   // over ET*8
    if (t >= (long long)ET * HEADS1) return;
    int h = (int)(t & 7), ei = (int)(t >> 3);
    int s, d; edge_sd(eidx, E, ei, s, d);
    float e = lrelu(as1[s * HEADS1 + h] + ad1[d * HEADS1 + h]);
    atomicMax(&mx1[d * HEADS1 + h], e);
}

__global__ void k_edge_denom1(const int* __restrict__ eidx, int E, int ET, int N,
                              const float* __restrict__ as1, const float* __restrict__ ad1,
                              const float* __restrict__ mx1, float* __restrict__ dn1) {
    long long t = (long long)blockIdx.x * blockDim.x + threadIdx.x;
    if (t >= (long long)ET * HEADS1) return;
    int h = (int)(t & 7), ei = (int)(t >> 3);
    int s, d; edge_sd(eidx, E, ei, s, d);
    float e = lrelu(as1[s * HEADS1 + h] + ad1[d * HEADS1 + h]);
    atomicAdd(&dn1[d * HEADS1 + h], expf(e - mx1[d * HEADS1 + h]));
}

__global__ void k_edge_aggr1(const int* __restrict__ eidx, int E, int ET, int N,
                             const float* __restrict__ as1, const float* __restrict__ ad1,
                             const float* __restrict__ mx1, const float* __restrict__ dn1,
                             const float* __restrict__ h1, float* __restrict__ out1) {
    long long t = (long long)blockIdx.x * blockDim.x + threadIdx.x;   // over ET*64
    if (t >= (long long)ET * F_HID) return;
    int f = (int)(t & 63), ei = (int)(t >> 6);
    int h = f >> 3;
    int s, d; edge_sd(eidx, E, ei, s, d);
    float e = lrelu(as1[s * HEADS1 + h] + ad1[d * HEADS1 + h]);
    float alpha = expf(e - mx1[d * HEADS1 + h]) / dn1[d * HEADS1 + h];
    atomicAdd(&out1[(size_t)d * F_HID + f], h1[(size_t)s * F_HID + f] * alpha);
}

__global__ void k_elu_bias1(float* __restrict__ out1, const float* __restrict__ b1, long long n) {
    long long i = (long long)blockIdx.x * blockDim.x + threadIdx.x;   // over N*64
    if (i >= n) return;
    float v = out1[i] + b1[i & 63];
    out1[i] = v > 0.f ? v : expm1f(v);                  // ELU (alpha=1); out1 becomes x2
}

// ---------------------------------------------------------------- layer 2 (64 -> 40, 1 head)
__global__ void k_gemm2(const float* __restrict__ x2, const float* __restrict__ W2,
                        float* __restrict__ h2, int N) {
    long long i = (long long)blockIdx.x * blockDim.x + threadIdx.x;   // over N*40
    if (i >= (long long)N * C_OUT) return;
    int n = (int)(i / C_OUT), c = (int)(i % C_OUT);
    const float* row = x2 + (size_t)n * F_HID;
    float s = 0.f;
    #pragma unroll
    for (int k = 0; k < F_HID; ++k) s += row[k] * W2[k * C_OUT + c];
    h2[i] = s;
}

__global__ void k_alpha2(const float* __restrict__ h2,
                         const float* __restrict__ att_s, const float* __restrict__ att_d,
                         float* __restrict__ as2, float* __restrict__ ad2, int N) {
    int n = blockIdx.x * blockDim.x + threadIdx.x;
    if (n >= N) return;
    const float* row = h2 + (size_t)n * C_OUT;
    float s = 0.f, d = 0.f;
    #pragma unroll
    for (int c = 0; c < C_OUT; ++c) { s += row[c] * att_s[c]; d += row[c] * att_d[c]; }
    as2[n] = s; ad2[n] = d;
}

__global__ void k_edge_max2(const int* __restrict__ eidx, int E, int ET,
                            const float* __restrict__ as2, const float* __restrict__ ad2,
                            float* __restrict__ mx2) {
    int ei = blockIdx.x * blockDim.x + threadIdx.x;
    if (ei >= ET) return;
    int s, d; edge_sd(eidx, E, ei, s, d);
    atomicMax(&mx2[d], lrelu(as2[s] + ad2[d]));
}

__global__ void k_edge_denom2(const int* __restrict__ eidx, int E, int ET,
                              const float* __restrict__ as2, const float* __restrict__ ad2,
                              const float* __restrict__ mx2, float* __restrict__ dn2) {
    int ei = blockIdx.x * blockDim.x + threadIdx.x;
    if (ei >= ET) return;
    int s, d; edge_sd(eidx, E, ei, s, d);
    atomicAdd(&dn2[d], expf(lrelu(as2[s] + ad2[d]) - mx2[d]));
}

__global__ void k_edge_aggr2(const int* __restrict__ eidx, int E, int ET,
                             const float* __restrict__ as2, const float* __restrict__ ad2,
                             const float* __restrict__ mx2, const float* __restrict__ dn2,
                             const float* __restrict__ h2, float* __restrict__ out) {
    long long t = (long long)blockIdx.x * blockDim.x + threadIdx.x;   // over ET*40
    if (t >= (long long)ET * C_OUT) return;
    int c = (int)(t % C_OUT), ei = (int)(t / C_OUT);
    int s, d; edge_sd(eidx, E, ei, s, d);
    float alpha = expf(lrelu(as2[s] + ad2[d]) - mx2[d]) / dn2[d];
    atomicAdd(&out[(size_t)d * C_OUT + c], h2[(size_t)s * C_OUT + c] * alpha);
}

__global__ void k_logsoftmax(float* __restrict__ out, const float* __restrict__ b2, int N) {
    int n = blockIdx.x * blockDim.x + threadIdx.x;
    if (n >= N) return;
    float v[C_OUT], m = -3.4e38f;
    #pragma unroll
    for (int c = 0; c < C_OUT; ++c) { v[c] = out[(size_t)n * C_OUT + c] + b2[c]; m = fmaxf(m, v[c]); }
    float sum = 0.f;
    #pragma unroll
    for (int c = 0; c < C_OUT; ++c) sum += expf(v[c] - m);
    float lg = logf(sum);
    #pragma unroll
    for (int c = 0; c < C_OUT; ++c) out[(size_t)n * C_OUT + c] = v[c] - m - lg;
}

// ---------------------------------------------------------------- launch
static inline int nblk(long long n, int b) { return (int)((n + b - 1) / b); }

extern "C" void kernel_launch(void* const* d_in, const int* in_sizes, int n_in,
                              void* d_out, int out_size, void* d_ws, size_t ws_size,
                              hipStream_t stream) {
    const float* x        = (const float*)d_in[0];
    const int*   eidx     = (const int*)d_in[1];
    const float* W1       = (const float*)d_in[2];
    const float* att_src1 = (const float*)d_in[3];
    const float* att_dst1 = (const float*)d_in[4];
    const float* b1       = (const float*)d_in[5];
    const float* W2       = (const float*)d_in[6];
    const float* att_src2 = (const float*)d_in[7];
    const float* att_dst2 = (const float*)d_in[8];
    const float* b2       = (const float*)d_in[9];

    const int N  = in_sizes[0] / F_IN;
    const int E  = in_sizes[1] / 2;
    const int ET = E + N;                         // + self loops
    float* out = (float*)d_out;                   // [N, 40], reused as layer-2 accumulator

    // workspace carve-out
    char* wp = (char*)d_ws;
    _Float16* W1p = (_Float16*)wp;  wp += (size_t)F_IN * F_HID * sizeof(_Float16);
    float* h1   = (float*)wp;  wp += (size_t)N * F_HID  * sizeof(float);
    float* out1 = (float*)wp;  wp += (size_t)N * F_HID  * sizeof(float);   // -> x2 after ELU
    float* as1  = (float*)wp;  wp += (size_t)N * HEADS1 * sizeof(float);
    float* ad1  = (float*)wp;  wp += (size_t)N * HEADS1 * sizeof(float);
    float* mx1  = (float*)wp;  wp += (size_t)N * HEADS1 * sizeof(float);
    float* dn1  = (float*)wp;  wp += (size_t)N * HEADS1 * sizeof(float);
    float* h2   = (float*)wp;  wp += (size_t)N * C_OUT  * sizeof(float);
    float* as2  = (float*)wp;  wp += (size_t)N * sizeof(float);
    float* ad2  = (float*)wp;  wp += (size_t)N * sizeof(float);
    float* mx2  = (float*)wp;  wp += (size_t)N * sizeof(float);
    float* dn2  = (float*)wp;  wp += (size_t)N * sizeof(float);

    const int B = 256;
    const float NEG_BIG = -3.4e38f;

    // init accumulators (every call: deterministic under graph replay)
    k_fill<<<nblk((long long)N * HEADS1, B), B, 0, stream>>>(mx1, NEG_BIG, (long long)N * HEADS1);
    k_fill<<<nblk((long long)N * HEADS1, B), B, 0, stream>>>(dn1, 0.f,     (long long)N * HEADS1);
    k_fill<<<nblk((long long)N * F_HID,  B), B, 0, stream>>>(out1, 0.f,    (long long)N * F_HID);
    k_fill<<<nblk(N, B), B, 0, stream>>>(mx2, NEG_BIG, N);
    k_fill<<<nblk(N, B), B, 0, stream>>>(dn2, 0.f, N);
    k_fill<<<nblk((long long)N * C_OUT, B), B, 0, stream>>>(out, 0.f, (long long)N * C_OUT);

    // layer 1
    k_pack_w1<<<nblk(F_IN * F_HID, B), B, 0, stream>>>(W1, W1p, F_IN * F_HID);
    k_gemm1_wmma<<<nblk(N, 32), 128, 0, stream>>>(x, W1p, h1, N);
    k_alpha1<<<nblk((long long)N * HEADS1, B), B, 0, stream>>>(h1, att_src1, att_dst1, as1, ad1, N);
    k_edge_max1  <<<nblk((long long)ET * HEADS1, B), B, 0, stream>>>(eidx, E, ET, N, as1, ad1, mx1);
    k_edge_denom1<<<nblk((long long)ET * HEADS1, B), B, 0, stream>>>(eidx, E, ET, N, as1, ad1, mx1, dn1);
    k_edge_aggr1 <<<nblk((long long)ET * F_HID,  B), B, 0, stream>>>(eidx, E, ET, N, as1, ad1, mx1, dn1, h1, out1);
    k_elu_bias1<<<nblk((long long)N * F_HID, B), B, 0, stream>>>(out1, b1, (long long)N * F_HID);

    // layer 2
    k_gemm2<<<nblk((long long)N * C_OUT, B), B, 0, stream>>>(out1, W2, h2, N);
    k_alpha2<<<nblk(N, B), B, 0, stream>>>(h2, att_src2, att_dst2, as2, ad2, N);
    k_edge_max2  <<<nblk(ET, B), B, 0, stream>>>(eidx, E, ET, as2, ad2, mx2);
    k_edge_denom2<<<nblk(ET, B), B, 0, stream>>>(eidx, E, ET, as2, ad2, mx2, dn2);
    k_edge_aggr2 <<<nblk((long long)ET * C_OUT, B), B, 0, stream>>>(eidx, E, ET, as2, ad2, mx2, dn2, h2, out);
    k_logsoftmax<<<nblk(N, B), B, 0, stream>>>(out, b2, N);
}
